// CapsuleLayer_37005438222848
// MI455X (gfx1250) — matrix operand order repaired and checked by reading, not compiled, not run on hardware
//
#include <hip/hip_runtime.h>

// ---------------------------------------------------------------------------
// Capsule dynamic routing, fused/recompute strategy for MI455X (gfx1250).
//   x: [B=32, I=2048, K=16] f32      W: [J=64, I=2048, D=32, K=16] f32
//   out v: [B=32, J=64, D=32] f32
// u_hat (512MB) never materialized; every routing pass recomputes it from W
// with v_wmma_f32_16x16x32_f16.  HBM traffic ~5 x 256MB (W) ~= 1.3GB,
// memory-bound at 23.3 TB/s => split-K over i for parallelism (512 blocks)
// and double-buffered, software-pipelined LDS staging to hide load latency.
// ---------------------------------------------------------------------------

namespace caps {
constexpr int B = 32, I = 2048, K = 16, J = 64, D = 32;
constexpr int SPLIT = 8;                     // split-K chunks over i
constexpr int ISTEP = 8;                     // i's staged per pipeline step
constexpr int NSTEP = (I / SPLIT) / ISTEP;   // 32 steps per block
}

typedef __attribute__((ext_vector_type(16))) _Float16 v16h;
typedef __attribute__((ext_vector_type(8)))  _Float16 h8;
typedef __attribute__((ext_vector_type(8)))  float    v8f;

#define SHUF16(a, b) \
  __builtin_shufflevector((a), (b), 0,1,2,3,4,5,6,7,8,9,10,11,12,13,14,15)

// ---------------------------------------------------------------------------
__global__ void zero_f32(float* __restrict__ p, int n) {
  int i = blockIdx.x * blockDim.x + threadIdx.x;
  int stride = gridDim.x * blockDim.x;
  for (; i < n; i += stride) p[i] = 0.0f;
}

// ---------------------------------------------------------------------------
// c[b,j,i] = softmax_j(bb[b,j,i]).  One thread per (b,i), loops over j=64.
__global__ void softmax_j(const float* __restrict__ bb, float* __restrict__ cc) {
  using namespace caps;
  int t = blockIdx.x * blockDim.x + threadIdx.x;
  if (t >= B * I) return;
  int b = t / I, i = t % I;
  const float* p = bb + (size_t)b * J * I + i;
  float m = -1e30f;
  for (int j = 0; j < J; ++j) m = fmaxf(m, p[(size_t)j * I]);
  float sum = 0.0f;
  for (int j = 0; j < J; ++j) sum += __expf(p[(size_t)j * I] - m);
  float inv = 1.0f / sum;
  float* q = cc + (size_t)b * J * I + i;
  for (int j = 0; j < J; ++j) q[(size_t)j * I] = __expf(p[(size_t)j * I] - m) * inv;
}

// ---------------------------------------------------------------------------
// Partial s:  sp[chunk][b,j,d] = sum_{i in chunk} c[b,j,i] * u_hat[b,j,i,d]
// computed as S_j[b,d] = (c*x)[b,(i,k)] @ W_j[(i,k),d], M=32,N=32,Kc=256*16.
// Grid: J*SPLIT blocks, 4 waves each (one 16x16 WMMA tile per wave).
// Software-pipelined: global loads for step n+1 issue between the LDS store
// and barrier of step n; double-buffered LDS, one barrier per step.
__global__ __launch_bounds__(128) void s_kernel(const float* __restrict__ x,
                                                const float* __restrict__ Wg,
                                                const float* __restrict__ cc,
                                                float* __restrict__ sp) {
  using namespace caps;
  __shared__ __align__(16) _Float16 lA[2][32][136]; // [buf][b][kk] (+pad)
  __shared__ __align__(16) _Float16 lB[2][32][136]; // [buf][d][kk]

  const int j     = blockIdx.x / SPLIT;
  const int chunk = blockIdx.x % SPLIT;
  const int ibase = chunk * (I / SPLIT);
  const int tid   = threadIdx.x;
  const int wave  = tid >> 5;
  const int lane  = tid & 31;
  const int mt    = wave >> 1;         // b-tile 0..1
  const int nt    = wave & 1;          // d-tile 0..1
  const int half  = lane >> 4;
  const int mrow  = mt * 16 + (lane & 15);
  const int ncol  = nt * 16 + (lane & 15);

  // staging registers (per thread: 2 (b,il) pairs for A, 2 (d,il) pairs for B)
  float  aC[2];
  float4 aX[2][4];
  float4 bW[2][4];

  auto loadStage = [&](int i0) {
#pragma unroll
    for (int p = 0; p < 2; ++p) {
      int e = tid + p * 128;
      {
        int il = e & 7, b = e >> 3;
        aC[p] = cc[((size_t)b * J + j) * I + i0 + il];
        const float4* xp = (const float4*)(x + ((size_t)b * I + i0 + il) * K);
        aX[p][0] = xp[0]; aX[p][1] = xp[1]; aX[p][2] = xp[2]; aX[p][3] = xp[3];
      }
      {
        int d = e & 31, il = e >> 5;
        const float4* wp =
            (const float4*)(Wg + (((size_t)j * I + i0 + il) * D + d) * K);
        bW[p][0] = wp[0]; bW[p][1] = wp[1]; bW[p][2] = wp[2]; bW[p][3] = wp[3];
      }
    }
  };

  auto storeStage = [&](int buf) {
#pragma unroll
    for (int p = 0; p < 2; ++p) {
      int e = tid + p * 128;
      {
        int il = e & 7, b = e >> 3;
        float c = aC[p];
        _Float16* dst = &lA[buf][b][il * 16];
        dst[0]  = (_Float16)(c * aX[p][0].x); dst[1]  = (_Float16)(c * aX[p][0].y);
        dst[2]  = (_Float16)(c * aX[p][0].z); dst[3]  = (_Float16)(c * aX[p][0].w);
        dst[4]  = (_Float16)(c * aX[p][1].x); dst[5]  = (_Float16)(c * aX[p][1].y);
        dst[6]  = (_Float16)(c * aX[p][1].z); dst[7]  = (_Float16)(c * aX[p][1].w);
        dst[8]  = (_Float16)(c * aX[p][2].x); dst[9]  = (_Float16)(c * aX[p][2].y);
        dst[10] = (_Float16)(c * aX[p][2].z); dst[11] = (_Float16)(c * aX[p][2].w);
        dst[12] = (_Float16)(c * aX[p][3].x); dst[13] = (_Float16)(c * aX[p][3].y);
        dst[14] = (_Float16)(c * aX[p][3].z); dst[15] = (_Float16)(c * aX[p][3].w);
      }
      {
        int d = e & 31, il = e >> 5;
        _Float16* dst = &lB[buf][d][il * 16];
        dst[0]  = (_Float16)bW[p][0].x; dst[1]  = (_Float16)bW[p][0].y;
        dst[2]  = (_Float16)bW[p][0].z; dst[3]  = (_Float16)bW[p][0].w;
        dst[4]  = (_Float16)bW[p][1].x; dst[5]  = (_Float16)bW[p][1].y;
        dst[6]  = (_Float16)bW[p][1].z; dst[7]  = (_Float16)bW[p][1].w;
        dst[8]  = (_Float16)bW[p][2].x; dst[9]  = (_Float16)bW[p][2].y;
        dst[10] = (_Float16)bW[p][2].z; dst[11] = (_Float16)bW[p][2].w;
        dst[12] = (_Float16)bW[p][3].x; dst[13] = (_Float16)bW[p][3].y;
        dst[14] = (_Float16)bW[p][3].z; dst[15] = (_Float16)bW[p][3].w;
      }
    }
  };

  v8f acc = {};
  loadStage(ibase);

  for (int step = 0; step < NSTEP; ++step) {
    const int buf = step & 1;
    storeStage(buf);
    if (step + 1 < NSTEP) {
      loadStage(ibase + (step + 1) * ISTEP);     // overlaps compute below
      if (step + 2 < NSTEP) {                    // L2 prefetch of W stream
        int d = tid & 31, il = tid >> 5;
        __builtin_prefetch(
            Wg + (((size_t)j * I + ibase + (step + 2) * ISTEP + il) * D + d) * K,
            0, 1);
      }
    }
    __syncthreads();
    // 4 WMMA k-steps of 32 over the 128 staged kk values
#pragma unroll
    for (int ks = 0; ks < 4; ++ks) {
      int cb = ks * 32;
      // A fragment (16x32): half0 -> K0..15, half1 -> K16..31
      h8 alo = *(const h8*)&lA[buf][mrow][cb + half * 16];
      h8 ahi = *(const h8*)&lA[buf][mrow][cb + half * 16 + 8];
      // B fragment (32x16): half0 -> K0..15, half1 -> K16..31, col = lane&15
      h8 blo = *(const h8*)&lB[buf][ncol][cb + half * 16];
      h8 bhi = *(const h8*)&lB[buf][ncol][cb + half * 16 + 8];
      v16h av = SHUF16(alo, ahi);
      v16h bv = SHUF16(blo, bhi);
      acc = __builtin_amdgcn_wmma_f32_16x16x32_f16(
          false, av, false, bv, (short)0, acc, false, false);
    }
  }

  // C/D layout: VGPR r -> M = r + 8*half, N = lane&15 (+tile offsets)
#pragma unroll
  for (int r = 0; r < 8; ++r) {
    int b = mt * 16 + r + 8 * half;
    sp[(size_t)chunk * (B * J * D) + ((size_t)b * J + j) * D + ncol] = acc[r];
  }
}

// ---------------------------------------------------------------------------
// v = squash( sum_chunk sp[chunk] ) over d.  One wave per (b,j), lane = d.
__global__ void squash_kernel(const float* __restrict__ sp, float* __restrict__ v) {
  using namespace caps;
  int wid  = (blockIdx.x * blockDim.x + threadIdx.x) >> 5;
  int lane = threadIdx.x & 31;
  if (wid >= B * J) return;
  float sv = 0.0f;
#pragma unroll
  for (int c = 0; c < SPLIT; ++c)
    sv += sp[(size_t)c * (B * J * D) + (size_t)wid * D + lane];
  float sq = sv * sv;
#pragma unroll
  for (int off = 16; off >= 1; off >>= 1) sq += __shfl_xor(sq, off, 32);
  float scale = sq / (1.0f + sq) * rsqrtf(sq + 1e-7f);
  v[(size_t)wid * D + lane] = scale * sv;
}

// ---------------------------------------------------------------------------
// bb[b,j,i] += sum_d v[b,j,d]*u_hat[b,j,i,d]
//   = sum_k x[b,i,k] * H[b,k],  H = V_j (32x32 over d) @ W[j,i] (32d x 16k).
// One wave per (j, 8 i's); Kc=32 = exactly one WMMA per (m-tile, i).
__global__ __launch_bounds__(256) void bupdate_kernel(const float* __restrict__ Wg,
                                                      const float* __restrict__ x,
                                                      const float* __restrict__ vv,
                                                      float* __restrict__ bb) {
  using namespace caps;
  __shared__ __align__(16) _Float16 lWt[8][16][40]; // per-wave [k][d] (pad 8)

  const int wv   = threadIdx.x >> 5;
  const int lane = threadIdx.x & 31;
  const int g    = blockIdx.x * 8 + wv;   // global wave id
  const int j    = g >> 8;                // I/8 = 256 chunks per j
  const int ic   = g & 255;
  const int half = lane >> 4;
  const int kcol = lane & 15;

  // A fragments for both b-tiles: A = v[b, d] as 16x32 f16
  v16h afrag[2];
#pragma unroll
  for (int mt = 0; mt < 2; ++mt) {
    union { _Float16 h[16]; v16h v; } u;
    int m = mt * 16 + kcol;
    int dbase = half * 16;                // half0: d0..15, half1: d16..31
    const float* vp = vv + ((size_t)m * J + j) * D;
#pragma unroll
    for (int e = 0; e < 8; ++e) u.h[e] = (_Float16)vp[dbase + e];
#pragma unroll
    for (int e = 0; e < 8; ++e) u.h[8 + e] = (_Float16)vp[dbase + 8 + e];
    afrag[mt] = u.v;
  }

  for (int ii = 0; ii < 8; ++ii) {
    int i = ic * 8 + ii;
    // stage W[j,i, d=lane, 0..15] transposed -> lWt[wv][k][lane] (f16)
    const float4* wp =
        (const float4*)(Wg + (((size_t)j * I + i) * D + lane) * K);
    float4 w0 = wp[0], w1 = wp[1], w2 = wp[2], w3 = wp[3];
    if (ii + 1 < 8)                       // L2 prefetch of next W tile
      __builtin_prefetch(Wg + (((size_t)j * I + i + 1) * D + lane) * K, 0, 1);
    _Float16 wt[16] = {
        (_Float16)w0.x, (_Float16)w0.y, (_Float16)w0.z, (_Float16)w0.w,
        (_Float16)w1.x, (_Float16)w1.y, (_Float16)w1.z, (_Float16)w1.w,
        (_Float16)w2.x, (_Float16)w2.y, (_Float16)w2.z, (_Float16)w2.w,
        (_Float16)w3.x, (_Float16)w3.y, (_Float16)w3.z, (_Float16)w3.w};
#pragma unroll
    for (int k = 0; k < 16; ++k) lWt[wv][k][lane] = wt[k];
    // wave-synchronous: LDS ops of one wave are in-order; no barrier needed.

    // B fragment: column k = kcol, contraction rows d = half*16 .. +15
    h8 blo = *(const h8*)&lWt[wv][kcol][half * 16];
    h8 bhi = *(const h8*)&lWt[wv][kcol][half * 16 + 8];
    v16h bv = SHUF16(blo, bhi);

#pragma unroll
    for (int mt = 0; mt < 2; ++mt) {
      v8f acc = {};
      acc = __builtin_amdgcn_wmma_f32_16x16x32_f16(
          false, afrag[mt], false, bv, (short)0, acc, false, false);
#pragma unroll
      for (int r = 0; r < 8; ++r) {
        int b = mt * 16 + r + 8 * half;  // uniform within each 16-lane group
        float red = acc[r] * x[((size_t)b * I + i) * K + kcol];
        red += __shfl_xor(red, 1, 32);
        red += __shfl_xor(red, 2, 32);
        red += __shfl_xor(red, 4, 32);
        red += __shfl_xor(red, 8, 32);
        if ((lane & 15) == 0)            // lane0 -> m=r, lane16 -> m=r+8
          bb[((size_t)b * J + j) * I + i] += red;
      }
    }
  }
}

// ---------------------------------------------------------------------------
extern "C" void kernel_launch(void* const* d_in, const int* in_sizes, int n_in,
                              void* d_out, int out_size, void* d_ws, size_t ws_size,
                              hipStream_t stream) {
  using namespace caps;
  const float* x = (const float*)d_in[0];
  const float* W = (const float*)d_in[1];
  float* v = (float*)d_out;

  // scratch (floats): bb [B*J*I] | cc [B*J*I] | sp [SPLIT*B*J*D]  (~34 MB)
  float* bb = (float*)d_ws;
  float* cc = bb + (size_t)B * J * I;
  float* sp = cc + (size_t)B * J * I;

  zero_f32<<<512, 256, 0, stream>>>(bb, B * J * I);

  for (int r = 0; r < 3; ++r) {
    softmax_j<<<(B * I + 255) / 256, 256, 0, stream>>>(bb, cc);
    s_kernel<<<J * SPLIT, 128, 0, stream>>>(x, W, cc, sp);
    squash_kernel<<<(B * J * 32 + 127) / 128, 128, 0, stream>>>(sp, v);
    if (r < 2)
      bupdate_kernel<<<(J * (I / 8)) / 8, 256, 0, stream>>>(W, x, v, bb);
  }
}
